// Encoder_38165079392904
// MI455X (gfx1250) — matrix operand order repaired and checked by reading, not compiled, not run on hardware
//
#include <hip/hip_runtime.h>
#include <hip/hip_bf16.h>

// ---------------------------------------------------------------------------
// ACT universal-transformer encoder for MI455X (gfx1250, wave32, WMMA).
// B=8 S=1024 D=512 H=8 DFF=2048 STEPS=6.  All GEMM/attention contractions run
// through v_wmma_f32_16x16x32_bf16 with f32 accumulation; activations/weights
// are converted to bf16 once per use.  Attention is flash-style (online
// softmax) so S x S logits never hit HBM.
// GEMM: 128x128 block tile, 8 waves each computing 32x64 via 2x4 WMMA accs
// (8 wmma per 32-K step; A-fragments reused across 4 B-fragments).
// ---------------------------------------------------------------------------

typedef __attribute__((ext_vector_type(16))) __bf16        v16bf;
typedef __attribute__((ext_vector_type(8)))  float         v8f;
typedef __attribute__((ext_vector_type(8)))  unsigned int  v8u;

#define B_   8
#define S_   1024
#define D_   512
#define H_   8
#define DH_  64
#define DFF_ 2048
#define STEPS_ 6
#define BS_  (B_ * S_)            // 8192 rows

__device__ __forceinline__ __bf16 f2bf(float f) {
  union { float f; unsigned u; } a; a.f = f;
  unsigned r = a.u + 0x7FFFu + ((a.u >> 16) & 1u);   // RNE
  unsigned short h = (unsigned short)(r >> 16);
  return __builtin_bit_cast(__bf16, h);
}

// ---------------------------------------------------------------------------
// Generic f32 -> bf16 convert
// ---------------------------------------------------------------------------
__global__ void cvt_f32_bf16(const float* __restrict__ src, __bf16* __restrict__ dst, int n) {
  int i = blockIdx.x * blockDim.x + threadIdx.x;
  if (i < n) dst[i] = f2bf(src[i]);
}

__global__ void zero_f32(float* __restrict__ p, int n) {
  int i = blockIdx.x * blockDim.x + threadIdx.x;
  if (i < n) p[i] = 0.0f;
}

// ---------------------------------------------------------------------------
// bf16 WMMA GEMM:  C[M,N] = A[M,K] * W[K,N] + bias (+Res) (relu?)
// Block: 256 threads (8 waves), tile 128x128; wave -> 32x64 via 2x4 wmma accs.
// flags: 1=relu, 2=add residual, 4=write f32, 8=write bf16
// ---------------------------------------------------------------------------
__global__ __launch_bounds__(256)
void gemm_bf16(const __bf16* __restrict__ A, const __bf16* __restrict__ W,
               const float* __restrict__ bias, const float* __restrict__ Res,
               float* __restrict__ Cf, __bf16* __restrict__ Cb,
               int M, int N, int K, int flags)
{
  __shared__ __bf16 Asm[128][34];    // +2 pad: kill LDS bank conflicts
  __shared__ __bf16 Wsm[128][34];    // W tile stored transposed [n][k]

  const int tid  = threadIdx.x;
  const int lane = tid & 31;
  const int wid  = tid >> 5;
  const int wm   = (wid >> 1) * 32;  // 0,32,64,96
  const int wn   = (wid & 1) * 64;   // 0,64
  const int lm   = lane & 15;
  const int kh   = lane >> 4;        // half-wave select

  const long rowBase = (long)blockIdx.y * 128;
  const long colBase = (long)blockIdx.x * 128;

  v8f acc[2][4];
#pragma unroll
  for (int mi = 0; mi < 2; ++mi)
#pragma unroll
    for (int ni = 0; ni < 4; ++ni) acc[mi][ni] = v8f{};

  for (int k0 = 0; k0 < K; k0 += 32) {
    __syncthreads();
    // ---- stage A tile 128x32 (row-major) -------------------------------
    {
      int r = tid >> 1;
      int c = (tid & 1) * 16;
      const unsigned* src = (const unsigned*)(A + (rowBase + r) * (long)K + k0 + c);
      unsigned* dst = (unsigned*)&Asm[r][c];
#pragma unroll
      for (int j = 0; j < 8; ++j) dst[j] = src[j];
    }
    // ---- stage W tile 32x128 transposed -> Wsm[n][k] -------------------
    {
      int kk = tid >> 3;               // 0..31
      int nc = (tid & 7) * 16;         // 0..112
      const __bf16* src = W + (long)(k0 + kk) * N + colBase + nc;
#pragma unroll
      for (int j = 0; j < 16; ++j) Wsm[nc + j][kk] = src[j];
    }
    if (k0 + 32 < K) {  // hint next tiles into cache (global_prefetch_b8)
      __builtin_prefetch(A + (rowBase + (tid >> 1)) * (long)K + k0 + 32, 0, 1);
      __builtin_prefetch(W + (long)(k0 + 32 + (tid >> 3)) * N + colBase, 0, 1);
    }
    __syncthreads();

    // ---- gather fragments (documented CDNA5 16-bit layouts) ------------
    v8u a0u, a1u;
    v8u bu[4];
#pragma unroll
    for (int i = 0; i < 8; ++i) {
      int ka = (i < 4 ? 2 * i : 16 + 2 * (i - 4)) + 8 * kh;  // A k-pattern
      int kb = (i < 4 ? 2 * i : 16 + 2 * (i - 4)) + 8 * kh;  // B k-pattern
      a0u[i] = *(const unsigned*)&Asm[wm + lm][ka];
      a1u[i] = *(const unsigned*)&Asm[wm + 16 + lm][ka];
#pragma unroll
      for (int ni = 0; ni < 4; ++ni)
        bu[ni][i] = *(const unsigned*)&Wsm[wn + ni * 16 + lm][kb];
    }
    v16bf a0 = __builtin_bit_cast(v16bf, a0u);
    v16bf a1 = __builtin_bit_cast(v16bf, a1u);
#pragma unroll
    for (int ni = 0; ni < 4; ++ni) {
      v16bf bf = __builtin_bit_cast(v16bf, bu[ni]);
      acc[0][ni] = __builtin_amdgcn_wmma_f32_16x16x32_bf16(false, a0, false, bf, (short)0, acc[0][ni], false, false);
      acc[1][ni] = __builtin_amdgcn_wmma_f32_16x16x32_bf16(false, a1, false, bf, (short)0, acc[1][ni], false, false);
    }
  }

  // ---- epilogue: bias / residual / relu / stores -----------------------
#pragma unroll
  for (int mi = 0; mi < 2; ++mi) {
#pragma unroll
    for (int ni = 0; ni < 4; ++ni) {
      long gcol = colBase + wn + ni * 16 + lm;
      float bv = bias[gcol];
#pragma unroll
      for (int r = 0; r < 8; ++r) {
        long grow = rowBase + wm + mi * 16 + 8 * kh + r;   // C layout: M = r (+8 for hi lanes)
        long idx  = grow * N + gcol;
        float v = acc[mi][ni][r] + bv;
        if (flags & 2) v += Res[idx];
        if (flags & 1) v = fmaxf(v, 0.0f);
        if (flags & 4) Cf[idx] = v;
        if (flags & 8) Cb[idx] = f2bf(v);
      }
    }
  }
}

// ---------------------------------------------------------------------------
// Flash attention: block = (b, h, 64 q-rows), 4 waves x 16 q-rows each.
// kv tiles of 32, online softmax, WMMA for QK^T and P*V.  dh = 64.
// Q,K,V,O are bf16 [B*S, D] with head h occupying columns h*64..h*64+63.
// ---------------------------------------------------------------------------
__global__ __launch_bounds__(128)
void attn_kernel(const __bf16* __restrict__ Q, const __bf16* __restrict__ Km,
                 const __bf16* __restrict__ V, __bf16* __restrict__ O)
{
  __shared__ __bf16 Ksm[32][66];       // [kv][dh]   (dh pairs contiguous)
  __shared__ __bf16 Vsm[64][34];       // [dh][kv]   (kv pairs contiguous)
  __shared__ __bf16 Pst[4][16][34];    // per-wave P staging (C->A layout fixup)

  const int tid  = threadIdx.x;
  const int lane = tid & 31;
  const int wid  = tid >> 5;
  const int lm   = lane & 15;
  const int kh   = lane >> 4;

  const int b = blockIdx.z, h = blockIdx.y;
  const int qbase = blockIdx.x * 64 + wid * 16;
  const long baseBH = (long)b * S_ * D_ + h * DH_;
  const float scale = 0.125f;          // 1/sqrt(64)

  // ---- Q fragments (A operand, 16 rows x dh=64 -> 2 fragments), load once
  v8u q0u, q1u;
  {
    const __bf16* qrow = Q + baseBH + (long)(qbase + lm) * D_;
#pragma unroll
    for (int i = 0; i < 8; ++i) {
      int ka = (i < 4 ? 2 * i : 16 + 2 * (i - 4)) + 8 * kh;
      q0u[i] = *(const unsigned*)(qrow + ka);
      q1u[i] = *(const unsigned*)(qrow + 32 + ka);
    }
  }
  v16bf qa0 = __builtin_bit_cast(v16bf, q0u);
  v16bf qa1 = __builtin_bit_cast(v16bf, q1u);

  v8f oacc[4] = {v8f{}, v8f{}, v8f{}, v8f{}};
  float mrow[8], lrow[8];
#pragma unroll
  for (int r = 0; r < 8; ++r) { mrow[r] = -3.0e38f; lrow[r] = 0.0f; }

  for (int kt = 0; kt < S_; kt += 32) {
    __syncthreads();
    // ---- stage K tile [32 kv][64 dh] -----------------------------------
    {
      int r = tid >> 2;
      int c = (tid & 3) * 16;
      const unsigned* src = (const unsigned*)(Km + baseBH + (long)(kt + r) * D_ + c);
      unsigned* dst = (unsigned*)&Ksm[r][c];
#pragma unroll
      for (int j = 0; j < 8; ++j) dst[j] = src[j];
    }
    // ---- stage V tile transposed -> Vsm[dh][kv] ------------------------
    {
      int r = tid >> 2;
      int c0 = (tid & 3) * 16;
      const __bf16* src = V + baseBH + (long)(kt + r) * D_ + c0;
#pragma unroll
      for (int j = 0; j < 16; ++j) Vsm[c0 + j][r] = src[j];
    }
    __syncthreads();

    // ---- QK^T: logits tile 16 q x 32 kv --------------------------------
    v8u b00u, b01u, b10u, b11u;
#pragma unroll
    for (int i = 0; i < 8; ++i) {
      int kb = (i < 4 ? 2 * i : 16 + 2 * (i - 4)) + 8 * kh;  // B k-pattern over dh chunk
      b00u[i] = *(const unsigned*)&Ksm[lm][kb];            // kv 0..15, dh 0..31
      b10u[i] = *(const unsigned*)&Ksm[lm][32 + kb];       // kv 0..15, dh 32..63
      b01u[i] = *(const unsigned*)&Ksm[16 + lm][kb];       // kv 16..31, dh 0..31
      b11u[i] = *(const unsigned*)&Ksm[16 + lm][32 + kb];  // kv 16..31, dh 32..63
    }
    v8f s0 = {}, s1 = {};
    s0 = __builtin_amdgcn_wmma_f32_16x16x32_bf16(false, qa0, false, __builtin_bit_cast(v16bf, b00u), (short)0, s0, false, false);
    s0 = __builtin_amdgcn_wmma_f32_16x16x32_bf16(false, qa1, false, __builtin_bit_cast(v16bf, b10u), (short)0, s0, false, false);
    s1 = __builtin_amdgcn_wmma_f32_16x16x32_bf16(false, qa0, false, __builtin_bit_cast(v16bf, b01u), (short)0, s1, false, false);
    s1 = __builtin_amdgcn_wmma_f32_16x16x32_bf16(false, qa1, false, __builtin_bit_cast(v16bf, b11u), (short)0, s1, false, false);

    // ---- online softmax (row stats across the 16-lane group) -----------
#pragma unroll
    for (int r = 0; r < 8; ++r) {
      float t0 = s0[r] * scale, t1 = s1[r] * scale;
      float mx = fmaxf(t0, t1);
#pragma unroll
      for (int m = 8; m >= 1; m >>= 1) mx = fmaxf(mx, __shfl_xor(mx, m, 32));
      float mn = fmaxf(mrow[r], mx);
      float alpha = __expf(mrow[r] - mn);
      float p0 = __expf(t0 - mn), p1 = __expf(t1 - mn);
      float ps = p0 + p1;
#pragma unroll
      for (int m = 8; m >= 1; m >>= 1) ps += __shfl_xor(ps, m, 32);
      lrow[r] = lrow[r] * alpha + ps;
      mrow[r] = mn;
#pragma unroll
      for (int j = 0; j < 4; ++j) oacc[j][r] *= alpha;
      Pst[wid][8 * kh + r][lm]      = f2bf(p0);   // C layout -> staging
      Pst[wid][8 * kh + r][16 + lm] = f2bf(p1);
    }

    // ---- P fragment (A layout) + P*V into O ----------------------------
    v8u pu;
#pragma unroll
    for (int i = 0; i < 8; ++i) {
      int ka = (i < 4 ? 2 * i : 16 + 2 * (i - 4)) + 8 * kh;
      pu[i] = *(const unsigned*)&Pst[wid][lm][ka];
    }
    v16bf pa = __builtin_bit_cast(v16bf, pu);
#pragma unroll
    for (int j = 0; j < 4; ++j) {
      v8u vu;
#pragma unroll
      for (int i = 0; i < 8; ++i) {
        int kb = (i < 4 ? 2 * i : 16 + 2 * (i - 4)) + 8 * kh;  // B k-pattern over kv
        vu[i] = *(const unsigned*)&Vsm[j * 16 + lm][kb];
      }
      oacc[j] = __builtin_amdgcn_wmma_f32_16x16x32_bf16(false, pa, false, __builtin_bit_cast(v16bf, vu), (short)0, oacc[j], false, false);
    }
  }

  // ---- epilogue: normalize and store ctx -------------------------------
#pragma unroll
  for (int j = 0; j < 4; ++j) {
#pragma unroll
    for (int r = 0; r < 8; ++r) {
      int row = qbase + 8 * kh + r;
      int col = j * 16 + lm;
      O[(long)(b * S_ + row) * D_ + h * DH_ + col] = f2bf(oacc[j][r] / lrow[r]);
    }
  }
}

// ---------------------------------------------------------------------------
// Row LayerNorm: block per row; optional f32 and bf16 outputs.
// ---------------------------------------------------------------------------
__global__ __launch_bounds__(256)
void ln_kernel(const float* __restrict__ X, const float* __restrict__ g,
               const float* __restrict__ bb, float* __restrict__ outF,
               __bf16* __restrict__ outB, int N)
{
  __shared__ float rs[256], rs2[256];
  const long row = blockIdx.x;
  const float* x = X + row * N;
  float s = 0.0f, s2 = 0.0f;
  for (int c = threadIdx.x; c < N; c += 256) { float v = x[c]; s += v; s2 += v * v; }
  rs[threadIdx.x] = s; rs2[threadIdx.x] = s2;
  __syncthreads();
  for (int o = 128; o > 0; o >>= 1) {
    if (threadIdx.x < o) { rs[threadIdx.x] += rs[threadIdx.x + o]; rs2[threadIdx.x] += rs2[threadIdx.x + o]; }
    __syncthreads();
  }
  float mean = rs[0] / (float)N;
  float var  = rs2[0] / (float)N - mean * mean;
  float inv  = rsqrtf(var + 1e-6f);
  for (int c = threadIdx.x; c < N; c += 256) {
    float v = (x[c] - mean) * inv * g[c] + bb[c];
    if (outF) outF[row * N + c] = v;
    if (outB) outB[row * N + c] = f2bf(v);
  }
}

// ---------------------------------------------------------------------------
// ACT ponder: wave per row; p = sigmoid(st . act_w + act_b); halting update.
// ---------------------------------------------------------------------------
__global__ __launch_bounds__(128)
void act_kernel(const float* __restrict__ st, const float* __restrict__ aw,
                const float* __restrict__ ab, float* __restrict__ hp,
                float* __restrict__ rem, float* __restrict__ nup,
                float* __restrict__ uw)
{
  const int lane = threadIdx.x & 31;
  const int wid  = threadIdx.x >> 5;
  const long row = (long)blockIdx.x * 4 + wid;
  const float* x = st + row * D_;
  float s = 0.0f;
  for (int c = lane; c < D_; c += 32) s += x[c] * aw[c];
#pragma unroll
  for (int m = 16; m >= 1; m >>= 1) s += __shfl_xor(s, m, 32);
  if (lane == 0) {
    const float thr = 1.0f - 0.01f;
    float p = 1.0f / (1.0f + __expf(-(s + ab[0])));
    float hp0 = hp[row];
    float still = (hp0 < 1.0f) ? 1.0f : 0.0f;
    float cand = hp0 + p * still;
    float new_halted = (cand > thr) ? still : 0.0f;
    float still2 = (cand <= thr) ? still : 0.0f;
    float hp1 = hp0 + p * still2;
    float rem1 = rem[row] + new_halted * (1.0f - hp1);
    float hp2 = hp1 + new_halted * rem1;
    nup[row] += still2 + new_halted;
    uw[row]  = p * still2 + new_halted * rem1;
    hp[row]  = hp2;
    rem[row] = rem1;
  }
}

// st = new_st * uw + st * (1 - uw)
__global__ void blend_kernel(float* __restrict__ st, const float* __restrict__ nst,
                             const float* __restrict__ uw, int n)
{
  int i = blockIdx.x * blockDim.x + threadIdx.x;
  if (i < n) {
    float u = uw[i / D_];
    st[i] = nst[i] * u + st[i] * (1.0f - u);
  }
}

// ---------------------------------------------------------------------------
// Host orchestration
// ---------------------------------------------------------------------------
extern "C" void kernel_launch(void* const* d_in, const int* in_sizes, int n_in,
                              void* d_out, int out_size, void* d_ws, size_t ws_size,
                              hipStream_t stream)
{
  (void)in_sizes; (void)n_in; (void)out_size; (void)ws_size;

  const float* x     = (const float*)d_in[0];
  const float* wq    = (const float*)d_in[1];  const float* bq  = (const float*)d_in[2];
  const float* wk    = (const float*)d_in[3];  const float* bk  = (const float*)d_in[4];
  const float* wv    = (const float*)d_in[5];  const float* bv  = (const float*)d_in[6];
  const float* wo    = (const float*)d_in[7];  const float* bo  = (const float*)d_in[8];
  const float* w1    = (const float*)d_in[9];  const float* b1  = (const float*)d_in[10];
  const float* lnf_g = (const float*)d_in[11]; const float* lnf_b = (const float*)d_in[12];
  const float* w2    = (const float*)d_in[13]; const float* b2  = (const float*)d_in[14];
  const float* ln1_g = (const float*)d_in[15]; const float* ln1_b = (const float*)d_in[16];
  const float* ln2_g = (const float*)d_in[17]; const float* ln2_b = (const float*)d_in[18];
  const float* act_w = (const float*)d_in[19]; const float* act_b = (const float*)d_in[20];

  char* ws = (char*)d_ws;
  size_t off = 0;
  auto alloc = [&](size_t bytes) { size_t o = off; off = (off + bytes + 255) & ~(size_t)255; return o; };

  __bf16* WqB = (__bf16*)(ws + alloc((size_t)D_ * D_ * 2));
  __bf16* WkB = (__bf16*)(ws + alloc((size_t)D_ * D_ * 2));
  __bf16* WvB = (__bf16*)(ws + alloc((size_t)D_ * D_ * 2));
  __bf16* WoB = (__bf16*)(ws + alloc((size_t)D_ * D_ * 2));
  __bf16* W1B = (__bf16*)(ws + alloc((size_t)D_ * DFF_ * 2));
  __bf16* W2B = (__bf16*)(ws + alloc((size_t)DFF_ * D_ * 2));

  float*  stF   = (float*)(ws + alloc((size_t)BS_ * D_ * 4));
  __bf16* Xb    = (__bf16*)(ws + alloc((size_t)BS_ * D_ * 2));
  __bf16* Qb    = (__bf16*)(ws + alloc((size_t)BS_ * D_ * 2));
  __bf16* Kb    = (__bf16*)(ws + alloc((size_t)BS_ * D_ * 2));
  __bf16* Vb    = (__bf16*)(ws + alloc((size_t)BS_ * D_ * 2));
  __bf16* ctxB  = (__bf16*)(ws + alloc((size_t)BS_ * D_ * 2));
  float*  tmpF  = (float*)(ws + alloc((size_t)BS_ * D_ * 4));   // pre-LN1, reused pre-LN2
  float*  out1F = (float*)(ws + alloc((size_t)BS_ * D_ * 4));
  __bf16* out1B = (__bf16*)(ws + alloc((size_t)BS_ * D_ * 2));
  float*  hF    = (float*)(ws + alloc((size_t)BS_ * DFF_ * 4));
  __bf16* hB    = (__bf16*)(ws + alloc((size_t)BS_ * DFF_ * 2));
  float*  nstF  = (float*)(ws + alloc((size_t)BS_ * D_ * 4));
  float*  hp    = (float*)(ws + alloc((size_t)BS_ * 4));
  float*  rem   = (float*)(ws + alloc((size_t)BS_ * 4));
  float*  nup   = (float*)(ws + alloc((size_t)BS_ * 4));
  float*  uw    = (float*)(ws + alloc((size_t)BS_ * 4));

  const int nDD  = D_ * D_;
  const int nDF  = D_ * DFF_;
  const int nAct = BS_ * D_;

  // weights -> bf16 (once per launch; deterministic)
  cvt_f32_bf16<<<(nDD + 255) / 256, 256, 0, stream>>>(wq, WqB, nDD);
  cvt_f32_bf16<<<(nDD + 255) / 256, 256, 0, stream>>>(wk, WkB, nDD);
  cvt_f32_bf16<<<(nDD + 255) / 256, 256, 0, stream>>>(wv, WvB, nDD);
  cvt_f32_bf16<<<(nDD + 255) / 256, 256, 0, stream>>>(wo, WoB, nDD);
  cvt_f32_bf16<<<(nDF + 255) / 256, 256, 0, stream>>>(w1, W1B, nDF);
  cvt_f32_bf16<<<(nDF + 255) / 256, 256, 0, stream>>>(w2, W2B, nDF);

  // state <- x ; zero ACT accumulators
  hipMemcpyAsync(stF, x, (size_t)nAct * 4, hipMemcpyDeviceToDevice, stream);
  zero_f32<<<(BS_ + 255) / 256, 256, 0, stream>>>(hp, BS_);
  zero_f32<<<(BS_ + 255) / 256, 256, 0, stream>>>(rem, BS_);
  zero_f32<<<(BS_ + 255) / 256, 256, 0, stream>>>(nup, BS_);

  const dim3 gQKV(D_ / 128, BS_ / 128);     // (4, 64)
  const dim3 gFFN1(DFF_ / 128, BS_ / 128);  // (16, 64)
  const dim3 gAttn(S_ / 64, H_, B_);        // (16, 8, 8)

  for (int step = 0; step < STEPS_; ++step) {
    // ACT ponder weights from state at step start
    act_kernel<<<BS_ / 4, 128, 0, stream>>>(stF, act_w, act_b, hp, rem, nup, uw);

    // state -> bf16
    cvt_f32_bf16<<<(nAct + 255) / 256, 256, 0, stream>>>(stF, Xb, nAct);

    // Q, K, V projections (bf16 out)
    gemm_bf16<<<gQKV, 256, 0, stream>>>(Xb, WqB, bq, nullptr, nullptr, Qb, BS_, D_, D_, 8);
    gemm_bf16<<<gQKV, 256, 0, stream>>>(Xb, WkB, bk, nullptr, nullptr, Kb, BS_, D_, D_, 8);
    gemm_bf16<<<gQKV, 256, 0, stream>>>(Xb, WvB, bv, nullptr, nullptr, Vb, BS_, D_, D_, 8);

    // flash attention -> ctx (bf16)
    attn_kernel<<<gAttn, 128, 0, stream>>>(Qb, Kb, Vb, ctxB);

    // out1 = LN1(state + ctx @ Wo + bo)
    gemm_bf16<<<gQKV, 256, 0, stream>>>(ctxB, WoB, bo, stF, tmpF, nullptr, BS_, D_, D_, 2 | 4);
    ln_kernel<<<BS_, 256, 0, stream>>>(tmpF, ln1_g, ln1_b, out1F, out1B, D_);

    // h = LNf(relu(out1 @ W1 + b1))
    gemm_bf16<<<gFFN1, 256, 0, stream>>>(out1B, W1B, b1, nullptr, hF, nullptr, BS_, DFF_, D_, 1 | 4);
    ln_kernel<<<BS_, 256, 0, stream>>>(hF, lnf_g, lnf_b, nullptr, hB, DFF_);

    // new_st = LN2(out1 + h @ W2 + b2)
    gemm_bf16<<<gQKV, 256, 0, stream>>>(hB, W2B, b2, out1F, tmpF, nullptr, BS_, D_, DFF_, 2 | 4);
    ln_kernel<<<BS_, 256, 0, stream>>>(tmpF, ln2_g, ln2_b, nstF, nullptr, D_);

    // st = new_st * uw + st * (1 - uw)
    blend_kernel<<<(nAct + 255) / 256, 256, 0, stream>>>(stF, nstF, uw, nAct);
  }

  hipMemcpyAsync(d_out, stF, (size_t)nAct * 4, hipMemcpyDeviceToDevice, stream);
}
